// ConditionalFusionModel_50328426775229
// MI455X (gfx1250) — matrix operand order repaired and checked by reading, not compiled, not run on hardware
//
#include <hip/hip_runtime.h>

#define Nn 20000
#define Ee 160000
#define Gg 1000
#define Hh 3
#define ETot (Ee + Nn)
#define GP 1008          // G padded to multiple of 16
#define OUTLD 1418       // 128+128+2+319+841

#define CDIV(a, b) (((a) + (b) - 1) / (b))

typedef __attribute__((ext_vector_type(16))) __bf16 v16bf;
typedef __attribute__((ext_vector_type(8)))  float  v8f;

union Frag { v16bf v; uint4 q[2]; };

static __device__ __forceinline__ __bf16 f2bf(float f) {
  union { float f; unsigned u; } in; in.f = f;
  unsigned u = in.u;
  unsigned r = (u + 0x7FFFu + ((u >> 16) & 1u)) >> 16;   // RNE
  union { unsigned short s; __bf16 b; } out; out.s = (unsigned short)r;
  return out.b;
}

static __device__ __forceinline__ unsigned enc_f(float f) {
  int ib = __float_as_int(f);
  return (ib < 0) ? ~(unsigned)ib : ((unsigned)ib | 0x80000000u);
}
static __device__ __forceinline__ float dec_f(unsigned u) {
  unsigned b = (u & 0x80000000u) ? (u & 0x7FFFFFFFu) : ~u;
  return __uint_as_float(b);
}

// ---------------- packing / conversion ----------------

// src f32 [M x K] row-major -> dst bf16 [Mpad x Kpad], zero padded (row-major, A operand)
__global__ void pack_bf16(const float* __restrict__ src, __bf16* __restrict__ dst,
                          int M, int K, int Mpad, int Kpad) {
  int t = blockIdx.x * blockDim.x + threadIdx.x;
  if (t >= Mpad * Kpad) return;
  int m = t / Kpad, k = t % Kpad;
  float v = (m < M && k < K) ? src[(size_t)m * K + k] : 0.f;
  dst[t] = f2bf(v);
}

// weight f32 [K x D] -> bf16 in WMMA B-fragment order:
// dst[(((nt*KB + kb32)*32 + lane)*16 + e] = W[kb32*32 + (lane>>4)*16 + e][nt*16 + (lane&15)]
// so each lane's 16 B elements for one 16x16x32 step are 32 contiguous bytes.
__global__ void convw_frag(const float* __restrict__ src, __bf16* __restrict__ dst,
                           int K, int D, int Kpad, int Dpad) {
  int t = blockIdx.x * blockDim.x + threadIdx.x;
  if (t >= Kpad * Dpad) return;
  int e    = t & 15;
  int lane = (t >> 4) & 31;
  int blk  = t >> 9;               // (nt*KB + kb32)
  int KB   = Kpad >> 5;
  int kb32 = blk % KB;
  int nt   = blk / KB;
  int k = (kb32 << 5) + ((lane >> 4) << 4) + e;
  int d = (nt << 4) + (lane & 15);
  float v = (k < K && d < D) ? src[(size_t)k * D + d] : 0.f;
  dst[t] = f2bf(v);
}

// ---------------- WMMA GEMM: C[MxDp] = A[Mpad x Kp](bf16 row-major) @ WP(frag) + bias ----------------
__global__ void wmma_gemm_bf16(const __bf16* __restrict__ A, const __bf16* __restrict__ WP,
                               const float* __restrict__ bias, int biasLen,
                               float* __restrict__ C,
                               int Mtiles, int Mreal, int Kpad, int DoutPad) {
  const int lane = threadIdx.x & 31;
  const int wave = threadIdx.x >> 5;
  const int tilesN = DoutPad >> 4;
  const int KB = Kpad >> 5;
  const int wg = blockIdx.x * (blockDim.x >> 5) + wave;
  if (wg >= Mtiles * tilesN) return;            // wave-uniform exit: EXEC stays full
  const int mt = wg / tilesN, nt = wg % tilesN;
  const int mbase = mt << 4, nbase = nt << 4;
  const int half = lane >> 4;                    // 0: lanes 0-15, 1: lanes 16-31
  const int ncol = nbase + (lane & 15);
  const uint4* Aq = (const uint4*)(A + (size_t)(mbase + (lane & 15)) * Kpad);
  const uint4* Bq = (const uint4*)(WP + ((size_t)nt * KB * 32 + lane) * 16);
  v8f c = {};
#pragma unroll 2
  for (int kb32 = 0; kb32 < KB; ++kb32) {
    if (kb32 + 1 < KB) __builtin_prefetch(Aq + (kb32 << 2) + 4, 0, 0);
    Frag a, b;
    // A 16x32 layout: lane holds K = {half*16..half*16+15} of row mrow
    a.q[0] = Aq[(kb32 << 2) + 2 * half];
    a.q[1] = Aq[(kb32 << 2) + 2 * half + 1];
    // B fragment: 32 contiguous bytes per lane per K-block
    b.q[0] = Bq[kb32 * 64];
    b.q[1] = Bq[kb32 * 64 + 1];
    c = __builtin_amdgcn_wmma_f32_16x16x32_bf16(false, a.v, false, b.v, (short)0, c, false, false);
  }
  const float bv = (bias != nullptr && ncol < biasLen) ? bias[ncol] : 0.f;
#pragma unroll
  for (int r = 0; r < 8; ++r) {
    const int m = mbase + half * 8 + r;          // C layout: VGPR r -> M = r + half*8
    if (m < Mreal) C[(size_t)m * DoutPad + ncol] = c[r] + bv;
  }
}

// ---------------- GATv2 edge phase ----------------

// wave per (edge, head): logit = sum_c lrelu(xl[src]+xr[dst]) * att ; segment-max via encoded atomicMax
__global__ void edge_logit(const int* __restrict__ ei,
                           const float* __restrict__ xl, const float* __restrict__ xr,
                           const float* __restrict__ att, int D,
                           float* __restrict__ logit, unsigned* __restrict__ menc) {
  int w = (blockIdx.x * blockDim.x + threadIdx.x) >> 5;
  int lane = threadIdx.x & 31;
  if (w >= ETot * Hh) return;
  int e = w / Hh, h = w % Hh;
  int s, d;
  if (e < Ee) { s = ei[e]; d = ei[Ee + e]; } else { s = d = e - Ee; }
  const int HD = Hh * D;
  const float* pl = xl + (size_t)s * HD + h * D;
  const float* pr = xr + (size_t)d * HD + h * D;
  const float* pa = att + h * D;
  float acc = 0.f;
  for (int cc = lane; cc < D; cc += 32) {
    float v = pl[cc] + pr[cc];
    v = v > 0.f ? v : 0.2f * v;                 // leaky_relu(0.2)
    acc += v * pa[cc];
  }
  for (int off = 16; off > 0; off >>= 1) acc += __shfl_down(acc, off, 32);
  if (lane == 0) {
    logit[(size_t)e * Hh + h] = acc;
    atomicMax(&menc[(size_t)d * Hh + h], enc_f(acc));
  }
}

// thread per (edge, head): ex = exp(logit - max); segment-sum denominator
__global__ void edge_exp(const int* __restrict__ ei,
                         float* __restrict__ logit, const unsigned* __restrict__ menc,
                         float* __restrict__ den) {
  int t = blockIdx.x * blockDim.x + threadIdx.x;
  if (t >= ETot * Hh) return;
  int e = t / Hh, h = t % Hh;
  int d = (e < Ee) ? ei[Ee + e] : e - Ee;
  float ex = __expf(logit[t] - dec_f(menc[(size_t)d * Hh + h]));
  logit[t] = ex;
  atomicAdd(&den[(size_t)d * Hh + h], ex);
}

// wave per (edge, head): hacc[dst][c] += xl[src][h*D+c] * alpha / H
__global__ void edge_agg(const int* __restrict__ ei,
                         const float* __restrict__ xl, const float* __restrict__ ex,
                         const float* __restrict__ den, int D,
                         float* __restrict__ hacc) {
  int w = (blockIdx.x * blockDim.x + threadIdx.x) >> 5;
  int lane = threadIdx.x & 31;
  if (w >= ETot * Hh) return;
  int e = w / Hh, h = w % Hh;
  int s, d;
  if (e < Ee) { s = ei[e]; d = ei[Ee + e]; } else { s = d = e - Ee; }
  float alpha = ex[(size_t)e * Hh + h] / (den[(size_t)d * Hh + h] + 1e-16f);
  float scale = alpha * (1.f / Hh);
  const float* pl = xl + (size_t)s * (Hh * D) + h * D;
  for (int cc = lane; cc < D; cc += 32)
    atomicAdd(&hacc[(size_t)d * D + cc], pl[cc] * scale);
}

// add bias, accumulate BN sum/sumsq
__global__ void bias_stats(float* __restrict__ h, const float* __restrict__ b,
                           int D, float* __restrict__ stats) {
  int t = blockIdx.x * blockDim.x + threadIdx.x;
  if (t >= Nn * D) return;
  int c = t % D;
  float v = h[t] + b[c];
  h[t] = v;
  atomicAdd(&stats[c], v);
  atomicAdd(&stats[D + c], v * v);
}

// batchnorm + ELU, write f32 in place and bf16 copy for the next GEMM
__global__ void bn_elu(float* __restrict__ h, const float* __restrict__ stats,
                       const float* __restrict__ g, const float* __restrict__ be,
                       int D, __bf16* __restrict__ actOut) {
  int t = blockIdx.x * blockDim.x + threadIdx.x;
  if (t >= Nn * D) return;
  int c = t % D;
  float mu = stats[c] * (1.f / Nn);
  float var = stats[D + c] * (1.f / Nn) - mu * mu;
  float y = g[c] * (h[t] - mu) * rsqrtf(var + 1e-5f) + be[c];
  float el = y > 0.f ? y : (__expf(y) - 1.f);   // elu
  h[t] = el;
  if (actOut) actOut[t] = f2bf(el);
}

// ---------------- decoder-side small kernels ----------------

__global__ void pool_sum(const float* __restrict__ h, const int* __restrict__ batch,
                         float* __restrict__ pooled) {
  int t = blockIdx.x * blockDim.x + threadIdx.x;
  if (t >= Nn * 256) return;
  int n = t / 256, c = t % 256;
  atomicAdd(&pooled[(size_t)batch[n] * 256 + c], h[t]);
}

__global__ void copy_cols(const float* __restrict__ src, int ld, int ncols,
                          float* __restrict__ out, int colOff) {
  int t = blockIdx.x * blockDim.x + threadIdx.x;
  if (t >= Gg * ncols) return;
  int gi = t / ncols, j = t % ncols;
  out[(size_t)gi * OUTLD + colOff + j] = src[(size_t)gi * ld + j];
}

__global__ void silu_inplace(float* __restrict__ x, int n) {
  int t = blockIdx.x * blockDim.x + threadIdx.x;
  if (t >= n) return;
  float v = x[t];
  x[t] = v * (1.f / (1.f + __expf(-v)));
}

// p = silu_s1 @ pw2 + pb2   (K=64, Dout=2); also writes out cols [256,258)
__global__ void pgemm(const float* __restrict__ s1, const float* __restrict__ w,
                      const float* __restrict__ b, float* __restrict__ pf,
                      float* __restrict__ out) {
  int t = blockIdx.x * blockDim.x + threadIdx.x;
  if (t >= Gg * 2) return;
  int gi = t / 2, j = t % 2;
  float acc = b[j];
  for (int k = 0; k < 64; ++k) acc += s1[(size_t)gi * 64 + k] * w[k * 2 + j];
  pf[t] = acc;
  out[(size_t)gi * OUTLD + 256 + j] = acc;
}

// zc = concat(mu[128], p[2]) -> bf16 [GP x 160] zero padded
__global__ void build_zc(const float* __restrict__ mu, const float* __restrict__ pf,
                         __bf16* __restrict__ zc) {
  int t = blockIdx.x * blockDim.x + threadIdx.x;
  if (t >= GP * 160) return;
  int gi = t / 160, k = t % 160;
  float v = 0.f;
  if (gi < Gg) {
    if (k < 128) v = mu[(size_t)gi * 128 + k];
    else if (k < 130) v = pf[(size_t)gi * 2 + (k - 128)];
  }
  zc[t] = f2bf(v);
}

// layernorm(512) + silu, pack to bf16 [GP x 512]
__global__ void ln_silu_pack(const float* __restrict__ x, const float* __restrict__ g,
                             const float* __restrict__ b, __bf16* __restrict__ bf) {
  __shared__ float sh[256];
  int row = blockIdx.x, tid = threadIdx.x;
  if (row >= Gg) {
    for (int k = tid; k < 512; k += 256) bf[(size_t)row * 512 + k] = f2bf(0.f);
    return;
  }
  float s = 0.f, s2 = 0.f;
  for (int k = tid; k < 512; k += 256) {
    float v = x[(size_t)row * 512 + k];
    s += v; s2 += v * v;
  }
  sh[tid] = s; __syncthreads();
  for (int o = 128; o > 0; o >>= 1) { if (tid < o) sh[tid] += sh[tid + o]; __syncthreads(); }
  float S = sh[0]; __syncthreads();
  sh[tid] = s2; __syncthreads();
  for (int o = 128; o > 0; o >>= 1) { if (tid < o) sh[tid] += sh[tid + o]; __syncthreads(); }
  float S2 = sh[0];
  float mean = S * (1.f / 512.f);
  float var = S2 * (1.f / 512.f) - mean * mean;
  float inv = rsqrtf(var + 1e-5f);
  for (int k = tid; k < 512; k += 256) {
    float v = x[(size_t)row * 512 + k];
    float y = g[k] * (v - mean) * inv + b[k];
    y = y * (1.f / (1.f + __expf(-y)));          // silu
    bf[(size_t)row * 512 + k] = f2bf(y);
  }
}

// x_rec + symmetrized adj into out cols [258, 1418)
__global__ void assemble(const float* __restrict__ o2, float* __restrict__ out) {
  int t = blockIdx.x * blockDim.x + threadIdx.x;
  if (t >= Gg * 1160) return;
  int gi = t / 1160, j = t % 1160;
  const float* row = o2 + (size_t)gi * 1168;
  float v;
  if (j < 319) v = row[j];
  else {
    int t2 = j - 319, i = t2 / 29, jj = t2 % 29;
    v = 0.5f * (row[319 + i * 29 + jj] + row[319 + jj * 29 + i]);
  }
  out[(size_t)gi * OUTLD + 258 + j] = v;
}

// ---------------- host launcher ----------------

extern "C" void kernel_launch(void* const* d_in, const int* in_sizes, int n_in,
                              void* d_out, int out_size, void* d_ws, size_t ws_size,
                              hipStream_t stream) {
  const float* x   = (const float*)d_in[0];
  const int*   ei  = (const int*)d_in[1];
  const int*   bat = (const int*)d_in[2];
  const float* w1l = (const float*)d_in[3],  *w1r = (const float*)d_in[4];
  const float* a1  = (const float*)d_in[5],  *b1  = (const float*)d_in[6];
  const float* g1  = (const float*)d_in[7],  *be1 = (const float*)d_in[8];
  const float* w2l = (const float*)d_in[9],  *w2r = (const float*)d_in[10];
  const float* a2  = (const float*)d_in[11], *b2  = (const float*)d_in[12];
  const float* g2  = (const float*)d_in[13], *be2 = (const float*)d_in[14];
  const float* w3l = (const float*)d_in[15], *w3r = (const float*)d_in[16];
  const float* a3  = (const float*)d_in[17], *b3  = (const float*)d_in[18];
  const float* g3  = (const float*)d_in[19], *be3 = (const float*)d_in[20];
  const float* wmu = (const float*)d_in[21], *bmu = (const float*)d_in[22];
  const float* wlv = (const float*)d_in[23], *blv = (const float*)d_in[24];
  const float* pw1 = (const float*)d_in[25], *pb1 = (const float*)d_in[26];
  const float* pw2 = (const float*)d_in[27], *pb2 = (const float*)d_in[28];
  const float* dw0 = (const float*)d_in[29], *db0 = (const float*)d_in[30];
  const float* dw1 = (const float*)d_in[31], *db1 = (const float*)d_in[32];
  const float* lng = (const float*)d_in[33], *lnb = (const float*)d_in[34];
  const float* dw2 = (const float*)d_in[35], *db2 = (const float*)d_in[36];
  float* out = (float*)d_out;

  char* ws = (char*)d_ws;
  size_t off = 0;
  auto alloc = [&](size_t bytes) -> void* {
    void* p = ws + off;
    off = (off + bytes + 255) & ~((size_t)255);
    return p;
  };

  float*    XL    = (float*)alloc((size_t)Nn * 768 * 4);
  float*    XR    = (float*)alloc((size_t)Nn * 768 * 4);
  float*    HACC  = (float*)alloc((size_t)Nn * 256 * 4);
  float*    LOGIT = (float*)alloc((size_t)ETot * Hh * 4);
  unsigned* MENC  = (unsigned*)alloc((size_t)Nn * Hh * 4);
  float*    DEN   = (float*)alloc((size_t)Nn * Hh * 4);
  __bf16*   ACT   = (__bf16*)alloc((size_t)Nn * 128 * 2);
  float*    STATS = (float*)alloc(2 * 256 * 4);
  __bf16* WB1L = (__bf16*)alloc(32 * 192 * 2),  *WB1R = (__bf16*)alloc(32 * 192 * 2);
  __bf16* WB2L = (__bf16*)alloc(64 * 384 * 2),  *WB2R = (__bf16*)alloc(64 * 384 * 2);
  __bf16* WB3L = (__bf16*)alloc(128 * 768 * 2), *WB3R = (__bf16*)alloc(128 * 768 * 2);
  __bf16* WBMU = (__bf16*)alloc(256 * 128 * 2), *WBLV = (__bf16*)alloc(256 * 128 * 2);
  __bf16* WBP1 = (__bf16*)alloc(128 * 64 * 2);
  __bf16* WBD0 = (__bf16*)alloc(160 * 256 * 2);
  __bf16* WBD1 = (__bf16*)alloc(256 * 512 * 2);
  __bf16* WBD2 = (__bf16*)alloc(512 * 1168 * 2);
  float*  POOLED = (float*)alloc((size_t)Gg * 256 * 4);
  __bf16* PBF  = (__bf16*)alloc((size_t)GP * 256 * 2);
  float*  MUF  = (float*)alloc((size_t)GP * 128 * 4);
  float*  LVF  = (float*)alloc((size_t)GP * 128 * 4);
  __bf16* ZBF  = (__bf16*)alloc((size_t)GP * 128 * 2);
  float*  S1F  = (float*)alloc((size_t)GP * 64 * 4);
  float*  PF   = (float*)alloc((size_t)GP * 2 * 4);
  __bf16* ZCBF = (__bf16*)alloc((size_t)GP * 160 * 2);
  float*  HDF  = (float*)alloc((size_t)GP * 256 * 4);
  __bf16* HDBF = (__bf16*)alloc((size_t)GP * 256 * 2);
  float*  H1F  = (float*)alloc((size_t)GP * 512 * 4);
  __bf16* H1BF = (__bf16*)alloc((size_t)GP * 512 * 2);
  float*  O2F  = (float*)alloc((size_t)GP * 1168 * 4);

  auto convw = [&](const float* s, __bf16* d, int K, int D, int Kp, int Dp) {
    convw_frag<<<CDIV(Kp * Dp, 256), 256, 0, stream>>>(s, d, K, D, Kp, Dp);
  };
  convw(w1l, WB1L, 11, 192, 32, 192);   convw(w1r, WB1R, 11, 192, 32, 192);
  convw(w2l, WB2L, 64, 384, 64, 384);   convw(w2r, WB2R, 64, 384, 64, 384);
  convw(w3l, WB3L, 128, 768, 128, 768); convw(w3r, WB3R, 128, 768, 128, 768);
  convw(wmu, WBMU, 256, 128, 256, 128); convw(wlv, WBLV, 256, 128, 256, 128);
  convw(pw1, WBP1, 128, 64, 128, 64);
  convw(dw0, WBD0, 130, 256, 160, 256);
  convw(dw1, WBD1, 256, 512, 256, 512);
  convw(dw2, WBD2, 512, 1160, 512, 1168);

  auto gemm = [&](const __bf16* A, const __bf16* W, const float* bias, int biasLen,
                  float* C, int Mtiles, int Mreal, int Kp, int Dp) {
    int waves = Mtiles * (Dp >> 4);
    wmma_gemm_bf16<<<CDIV(waves, 8), 256, 0, stream>>>(A, W, bias, biasLen, C, Mtiles, Mreal, Kp, Dp);
  };

  auto run_layer = [&](const __bf16* WL, const __bf16* WR, const float* att,
                       const float* bb, const float* gg, const float* bbeta,
                       int Kp, int D, __bf16* actNext) {
    const int HD = Hh * D;
    gemm(ACT, WL, nullptr, 0, XL, Nn / 16, Nn, Kp, HD);
    gemm(ACT, WR, nullptr, 0, XR, Nn / 16, Nn, Kp, HD);
    hipMemsetAsync(MENC, 0, (size_t)Nn * Hh * 4, stream);
    hipMemsetAsync(DEN, 0, (size_t)Nn * Hh * 4, stream);
    hipMemsetAsync(HACC, 0, (size_t)Nn * D * 4, stream);
    hipMemsetAsync(STATS, 0, 2 * D * 4, stream);
    const int nw = ETot * Hh;
    edge_logit<<<CDIV(nw * 32, 256), 256, 0, stream>>>(ei, XL, XR, att, D, LOGIT, MENC);
    edge_exp<<<CDIV(nw, 256), 256, 0, stream>>>(ei, LOGIT, MENC, DEN);
    edge_agg<<<CDIV(nw * 32, 256), 256, 0, stream>>>(ei, XL, LOGIT, DEN, D, HACC);
    bias_stats<<<CDIV(Nn * D, 256), 256, 0, stream>>>(HACC, bb, D, STATS);
    bn_elu<<<CDIV(Nn * D, 256), 256, 0, stream>>>(HACC, STATS, gg, bbeta, D, actNext);
  };

  // encoder: 3 GATv2 layers
  pack_bf16<<<CDIV(Nn * 32, 256), 256, 0, stream>>>(x, ACT, Nn, 11, Nn, 32);
  run_layer(WB1L, WB1R, a1, b1, g1, be1, 32, 64, ACT);       // h1 [N,64]  -> ACT bf16
  run_layer(WB2L, WB2R, a2, b2, g2, be2, 64, 128, ACT);      // h2 [N,128] -> ACT bf16
  run_layer(WB3L, WB3R, a3, b3, g3, be3, 128, 256, nullptr); // h3 [N,256] in HACC (f32)

  // pooling
  hipMemsetAsync(POOLED, 0, (size_t)Gg * 256 * 4, stream);
  pool_sum<<<CDIV(Nn * 256, 256), 256, 0, stream>>>(HACC, bat, POOLED);
  pack_bf16<<<CDIV(GP * 256, 256), 256, 0, stream>>>(POOLED, PBF, Gg, 256, GP, 256);

  // decoder head
  gemm(PBF, WBMU, bmu, 128, MUF, GP / 16, Gg, 256, 128);
  gemm(PBF, WBLV, blv, 128, LVF, GP / 16, Gg, 256, 128);
  copy_cols<<<CDIV(Gg * 128, 256), 256, 0, stream>>>(MUF, 128, 128, out, 0);
  copy_cols<<<CDIV(Gg * 128, 256), 256, 0, stream>>>(LVF, 128, 128, out, 128);

  pack_bf16<<<CDIV(GP * 128, 256), 256, 0, stream>>>(MUF, ZBF, Gg, 128, GP, 128);
  gemm(ZBF, WBP1, pb1, 64, S1F, GP / 16, Gg, 128, 64);
  silu_inplace<<<CDIV(Gg * 64, 256), 256, 0, stream>>>(S1F, Gg * 64);
  pgemm<<<CDIV(Gg * 2, 256), 256, 0, stream>>>(S1F, pw2, pb2, PF, out);

  build_zc<<<CDIV(GP * 160, 256), 256, 0, stream>>>(MUF, PF, ZCBF);
  gemm(ZCBF, WBD0, db0, 256, HDF, GP / 16, Gg, 160, 256);
  pack_bf16<<<CDIV(GP * 256, 256), 256, 0, stream>>>(HDF, HDBF, Gg, 256, GP, 256);
  gemm(HDBF, WBD1, db1, 512, H1F, GP / 16, Gg, 256, 512);
  ln_silu_pack<<<GP, 256, 0, stream>>>(H1F, lng, lnb, H1BF);
  gemm(H1BF, WBD2, db2, 1160, O2F, GP / 16, Gg, 512, 1168);
  assemble<<<CDIV(Gg * 1160, 256), 256, 0, stream>>>(O2F, out);
}
